// AttnDecoderRNN_60455959658681
// MI455X (gfx1250) — matrix operand order repaired and checked by reading, not compiled, not run on hardware
//
#include <hip/hip_runtime.h>
#include <hip/hip_bf16.h>
#include <math.h>

#define Hdim 512
#define Vdim 50257
#define Bdim 256
#define Ldim 50

typedef __attribute__((ext_vector_type(2)))  float  v2f;
typedef __attribute__((ext_vector_type(8)))  float  v8f;
typedef __attribute__((ext_vector_type(16))) __bf16 v16bf;

// fp32 -> bf16 with round-to-nearest-even (pure integer, no FP cvt needed)
__device__ inline __bf16 f2bf(float f) {
    union { float f; unsigned u; } x; x.f = f;
    unsigned r = x.u + 0x7FFFu + ((x.u >> 16) & 1u);
    union { unsigned short s; __bf16 b; } y; y.s = (unsigned short)(r >> 16);
    return y.b;
}

// ---------------------------------------------------------------------------
// K1: embedding gather + build concat buffers
// ---------------------------------------------------------------------------
__global__ void prep_kernel(const int* __restrict__ tok,
                            const float* __restrict__ h0,
                            const float* __restrict__ emb,
                            float* __restrict__ cat_eh,
                            float* __restrict__ cat_ea) {
    int b = blockIdx.x;
    int t = threadIdx.x;
    int row = tok[b];
    for (int h = t; h < Hdim; h += 256) {
        float e = emb[(size_t)row * Hdim + h];
        cat_eh[(size_t)b * (2 * Hdim) + h]        = e;
        cat_eh[(size_t)b * (2 * Hdim) + Hdim + h] = h0[(size_t)b * Hdim + h];
        cat_ea[(size_t)b * (2 * Hdim) + h]        = e;
    }
}

// ---------------------------------------------------------------------------
// Generic fp32 WMMA GEMM:  out[MxN] = A[MxK] * W[NxK]^T + bias, optional ReLU
// One wave per 16x16 output tile (small/medium GEMMs; operands are L2-hot).
// ---------------------------------------------------------------------------
__global__ void gemm_f32_wmma(const float* __restrict__ A,
                              const float* __restrict__ W,
                              const float* __restrict__ bias,
                              float* __restrict__ out,
                              int N, int K, int relu) {
    int lane = threadIdx.x & 31;
    int half = lane >> 4;
    int m    = lane & 15;
    int m0   = blockIdx.y * 16;
    int n0   = blockIdx.x * 16;

    int nrow   = n0 + m;
    int nvalid = (nrow < N);
    const float2* Arow = (const float2*)(A + (size_t)(m0 + m) * K);
    const float2* Wrow = (const float2*)(W + (size_t)(nvalid ? nrow : 0) * K);
    float wscale = nvalid ? 1.0f : 0.0f;

    v8f acc = {};
#pragma unroll 4
    for (int kk = 0; kk < K; kk += 4) {
        int idx = (kk >> 1) + half;
        float2 af = Arow[idx];
        float2 wf = Wrow[idx];
        v2f a; a[0] = af.x;          a[1] = af.y;
        v2f b; b[0] = wf.x * wscale; b[1] = wf.y * wscale;
        acc = __builtin_amdgcn_wmma_f32_16x16x4_f32(
            false, a, false, b, (short)0, acc, false, false);
    }

    int col = n0 + m;
    if (col < N) {
        float bv = bias ? bias[col] : 0.0f;
#pragma unroll
        for (int r = 0; r < 8; ++r) {
            int row = m0 + half * 8 + r;
            float v = acc[r] + bv;
            if (relu) v = fmaxf(v, 0.0f);
            out[(size_t)row * N + col] = v;
        }
    }
}

// ---------------------------------------------------------------------------
// K3: row softmax over L=50 (one thread per row; tiny)
// ---------------------------------------------------------------------------
__global__ void softmax_rows(const float* __restrict__ logits,
                             float* __restrict__ w_ws,
                             float* __restrict__ w_out) {
    int b = threadIdx.x;
    const float* row = logits + (size_t)b * Ldim;
    float mx = row[0];
    for (int l = 1; l < Ldim; ++l) mx = fmaxf(mx, row[l]);
    float s = 0.0f;
    float e[Ldim];
    for (int l = 0; l < Ldim; ++l) { e[l] = expf(row[l] - mx); s += e[l]; }
    float inv = 1.0f / s;
    for (int l = 0; l < Ldim; ++l) {
        float v = e[l] * inv;
        w_ws [(size_t)b * Ldim + l] = v;
        w_out[(size_t)b * Ldim + l] = v;
    }
}

// ---------------------------------------------------------------------------
// K4: attn_applied[b,h] = sum_l w[b,l] * enc[l,b,h]  -> cat_ea[b, H + h]
// ---------------------------------------------------------------------------
__global__ void attn_apply(const float* __restrict__ attnw,
                           const float* __restrict__ enc,
                           float* __restrict__ cat_ea) {
    __shared__ float w[Ldim];
    int b = blockIdx.x;
    int t = threadIdx.x;
    if (t < Ldim) w[t] = attnw[(size_t)b * Ldim + t];
    __syncthreads();
    for (int h = t; h < Hdim; h += 256) {
        float acc = 0.0f;
        for (int l = 0; l < Ldim; ++l)
            acc += w[l] * enc[((size_t)l * Bdim + b) * Hdim + h];
        cat_ea[(size_t)b * (2 * Hdim) + Hdim + h] = acc;
    }
}

// ---------------------------------------------------------------------------
// K6: GRU elementwise combine (PyTorch gate order r,z,n)
// ---------------------------------------------------------------------------
__global__ void gru_elem(const float* __restrict__ gi,
                         const float* __restrict__ gh,
                         const float* __restrict__ h0,
                         float* __restrict__ h_new) {
    int idx = blockIdx.x * blockDim.x + threadIdx.x;
    if (idx >= Bdim * Hdim) return;
    int b = idx >> 9;
    int h = idx & (Hdim - 1);
    const float* gib = gi + (size_t)b * 3 * Hdim;
    const float* ghb = gh + (size_t)b * 3 * Hdim;
    float r = 1.0f / (1.0f + expf(-(gib[h]        + ghb[h])));
    float z = 1.0f / (1.0f + expf(-(gib[Hdim + h] + ghb[Hdim + h])));
    float n = tanhf(gib[2 * Hdim + h] + r * ghb[2 * Hdim + h]);
    float hp = h0[idx];
    h_new[idx] = (1.0f - z) * n + z * hp;
}

// ---------------------------------------------------------------------------
// K7: fc logits via bf16 WMMA, fp32 accumulate.
//   logits[b,v] = sum_h h_new[b,h]*fc_w[v,h] + fc_b[v]
//
// Key data-movement property: fc_w (103 MB fp32) is streamed from HBM exactly
// ONCE.  Each block (8 waves) covers the FULL M=256 batch for 8 v-tiles; the
// whole h_new matrix, converted to bf16 and pre-swizzled into the WMMA
// A-fragment layout, is cached in 256 KB of LDS (CDNA5: 320 KB/WGP).
//
// LDS layout: chunk(mt, kstep, lane) = 16 bf16 values (32 B), at byte offset
//   ((mt*16 + kstep)*32 + lane) * 32
// holding A[mt*16 + (lane&15)][kstep*32 + {h*8+0..7, 16+h*8+0..7}], h=lane>>4,
// i.e. exactly one lane's v16bf A-fragment -> two ds_load_b128 per fragment.
//
// Per wave: one 16-wide v-tile, 16 accumulators (128 VGPRs) over all m-tiles.
// ---------------------------------------------------------------------------
#define FC_WAVES 8
__global__ void fc_bf16_wmma(const float* __restrict__ hnew,
                             const float* __restrict__ fcw,
                             const float* __restrict__ fcb,
                             float* __restrict__ logits) {
    extern __shared__ __align__(16) __bf16 fc_lds[];   // 16*16*32*16 = 131072 bf16 = 256 KB

    int t = threadIdx.x;

    // ---- cooperative fill: h_new (fp32, L2-hot) -> swizzled bf16 fragments --
    for (int c = t; c < 16 * 16 * 32; c += 256) {      // 8192 lane-chunks
        int lane_c = c & 31;
        int kstep  = (c >> 5) & 15;
        int mt     = c >> 9;
        int half_c = lane_c >> 4;
        int m      = mt * 16 + (lane_c & 15);
        const float* src = hnew + (size_t)m * Hdim + kstep * 32 + half_c * 8;
        const float4* p0 = (const float4*)(src);
        const float4* p1 = (const float4*)(src + 16);
        float4 x0 = p0[0], x1 = p0[1];
        float4 x2 = p1[0], x3 = p1[1];
        __bf16* dst = fc_lds + (size_t)c * 16;
        dst[0]  = f2bf(x0.x); dst[1]  = f2bf(x0.y); dst[2]  = f2bf(x0.z); dst[3]  = f2bf(x0.w);
        dst[4]  = f2bf(x1.x); dst[5]  = f2bf(x1.y); dst[6]  = f2bf(x1.z); dst[7]  = f2bf(x1.w);
        dst[8]  = f2bf(x2.x); dst[9]  = f2bf(x2.y); dst[10] = f2bf(x2.z); dst[11] = f2bf(x2.w);
        dst[12] = f2bf(x3.x); dst[13] = f2bf(x3.y); dst[14] = f2bf(x3.z); dst[15] = f2bf(x3.w);
    }
    __syncthreads();

    // ---- compute: one v-tile per wave, all 16 m-tiles ----------------------
    int wave = t >> 5;
    int lane = t & 31;
    int half = lane >> 4;
    int l15  = lane & 15;

    int vt = blockIdx.x * FC_WAVES + wave;             // v-tile index
    int v  = vt * 16 + l15;
    int vvalid = (v < Vdim);
    float wsc  = vvalid ? 1.0f : 0.0f;
    const float* Wrow = fcw + (size_t)(vvalid ? v : (Vdim - 1)) * Hdim;

    v8f acc[16];
#pragma unroll
    for (int mt = 0; mt < 16; ++mt) acc[mt] = (v8f){};

    for (int kk = 0; kk < Hdim; kk += 32) {
        int kstep = kk >> 5;
        const float4* pw = (const float4*)(Wrow + kk + half * 8);
        float4 w0 = pw[0], w1 = pw[1], w2 = pw[4], w3 = pw[5];

        v16bf b;
        b[0]  = f2bf(w0.x * wsc); b[1]  = f2bf(w0.y * wsc);
        b[2]  = f2bf(w0.z * wsc); b[3]  = f2bf(w0.w * wsc);
        b[4]  = f2bf(w1.x * wsc); b[5]  = f2bf(w1.y * wsc);
        b[6]  = f2bf(w1.z * wsc); b[7]  = f2bf(w1.w * wsc);
        b[8]  = f2bf(w2.x * wsc); b[9]  = f2bf(w2.y * wsc);
        b[10] = f2bf(w2.z * wsc); b[11] = f2bf(w2.w * wsc);
        b[12] = f2bf(w3.x * wsc); b[13] = f2bf(w3.y * wsc);
        b[14] = f2bf(w3.z * wsc); b[15] = f2bf(w3.w * wsc);

        const __bf16* abase = fc_lds + (size_t)kstep * 512 + (size_t)lane * 16;
#pragma unroll
        for (int mt = 0; mt < 16; ++mt) {
            v16bf a = *(const v16bf*)(abase + (size_t)mt * 8192);
            acc[mt] = __builtin_amdgcn_wmma_f32_16x16x32_bf16(
                false, a, false, b, (short)0, acc[mt], false, false);
        }
    }

    if (vvalid) {
        float bv = fcb[v];
#pragma unroll
        for (int mt = 0; mt < 16; ++mt) {
#pragma unroll
            for (int r = 0; r < 8; ++r) {
                int row = mt * 16 + half * 8 + r;
                logits[(size_t)row * Vdim + v] = acc[mt][r] + bv;
            }
        }
    }
}

// ---------------------------------------------------------------------------
// K8: in-place log-softmax per row of V (block per row)
// ---------------------------------------------------------------------------
__global__ void logsoftmax_rows(float* __restrict__ out) {
    __shared__ float sd[256];
    int b = blockIdx.x;
    int t = threadIdx.x;
    float* row = out + (size_t)b * Vdim;

    float mx = -INFINITY;
    for (int i = t; i < Vdim; i += 256) mx = fmaxf(mx, row[i]);
    sd[t] = mx; __syncthreads();
    for (int s = 128; s > 0; s >>= 1) {
        if (t < s) sd[t] = fmaxf(sd[t], sd[t + s]);
        __syncthreads();
    }
    mx = sd[0]; __syncthreads();

    float sum = 0.0f;
    for (int i = t; i < Vdim; i += 256) sum += expf(row[i] - mx);
    sd[t] = sum; __syncthreads();
    for (int s = 128; s > 0; s >>= 1) {
        if (t < s) sd[t] += sd[t + s];
        __syncthreads();
    }
    float lse = mx + logf(sd[0]);
    __syncthreads();

    for (int i = t; i < Vdim; i += 256) row[i] -= lse;
}

// ---------------------------------------------------------------------------
extern "C" void kernel_launch(void* const* d_in, const int* in_sizes, int n_in,
                              void* d_out, int out_size, void* d_ws, size_t ws_size,
                              hipStream_t stream) {
    const int*   tok    = (const int*)  d_in[0];
    const float* hidden = (const float*)d_in[1];   // [1,B,H] -> [B,H]
    const float* enc    = (const float*)d_in[2];   // [L,B,H]
    const float* emb    = (const float*)d_in[3];   // [V,H]
    const float* attn_w = (const float*)d_in[4];   // [L,2H]
    const float* attn_b = (const float*)d_in[5];   // [L]
    const float* comb_w = (const float*)d_in[6];   // [H,2H]
    const float* comb_b = (const float*)d_in[7];   // [H]
    const float* w_ih   = (const float*)d_in[8];   // [3H,H]
    const float* w_hh   = (const float*)d_in[9];   // [3H,H]
    const float* b_ih   = (const float*)d_in[10];  // [3H]
    const float* b_hh   = (const float*)d_in[11];  // [3H]
    const float* fc_w   = (const float*)d_in[12];  // [V,H]
    const float* fc_b   = (const float*)d_in[13];  // [V]

    float* out_logp = (float*)d_out;                               // [B,V]
    float* out_h    = out_logp + (size_t)Bdim * Vdim;              // [B,H]
    float* out_attn = out_h    + (size_t)Bdim * Hdim;              // [B,L]

    // workspace carve-up (floats)
    float* ws      = (float*)d_ws;
    float* cat_eh  = ws;                                           // B*2H
    float* cat_ea  = cat_eh  + (size_t)Bdim * 2 * Hdim;            // B*2H
    float* attnlog = cat_ea  + (size_t)Bdim * 2 * Hdim;            // B*L
    float* attnw   = attnlog + (size_t)Bdim * Ldim;                // B*L
    float* xbuf    = attnw   + (size_t)Bdim * Ldim;                // B*H
    float* gi      = xbuf    + (size_t)Bdim * Hdim;                // B*3H
    float* gh      = gi      + (size_t)Bdim * 3 * Hdim;            // B*3H

    // 1) embedding + concat buffers
    prep_kernel<<<Bdim, 256, 0, stream>>>(tok, hidden, emb, cat_eh, cat_ea);

    // 2) attention scores: [B,2H]@[2H,L]+b  (fp32 WMMA)
    gemm_f32_wmma<<<dim3((Ldim + 15) / 16, Bdim / 16), 32, 0, stream>>>(
        cat_eh, attn_w, attn_b, attnlog, Ldim, 2 * Hdim, 0);

    // 3) softmax over L -> attn_weights (ws copy + output)
    softmax_rows<<<1, Bdim, 0, stream>>>(attnlog, attnw, out_attn);

    // 4) attn_applied -> cat_ea[:, H:2H]
    attn_apply<<<Bdim, 256, 0, stream>>>(attnw, enc, cat_ea);

    // 5) combine + ReLU: [B,2H]@[2H,H]+b  (fp32 WMMA)
    gemm_f32_wmma<<<dim3(Hdim / 16, Bdim / 16), 32, 0, stream>>>(
        cat_ea, comb_w, comb_b, xbuf, Hdim, 2 * Hdim, 1);

    // 6) GRU gates (fp32 WMMA) + elementwise
    gemm_f32_wmma<<<dim3(3 * Hdim / 16, Bdim / 16), 32, 0, stream>>>(
        xbuf, w_ih, b_ih, gi, 3 * Hdim, Hdim, 0);
    gemm_f32_wmma<<<dim3(3 * Hdim / 16, Bdim / 16), 32, 0, stream>>>(
        hidden, w_hh, b_hh, gh, 3 * Hdim, Hdim, 0);
    gru_elem<<<(Bdim * Hdim + 255) / 256, 256, 0, stream>>>(gi, gh, hidden, out_h);

    // 7) fc logits (bf16 WMMA): fc_w streamed from HBM exactly once,
    //    h_new cached in 256 KB LDS as pre-swizzled bf16 A-fragments.
    int vtiles = (Vdim + 15) / 16;                                 // 3142
    fc_bf16_wmma<<<dim3((vtiles + FC_WAVES - 1) / FC_WAVES), 256,
                   256 * 1024, stream>>>(out_h, fc_w, fc_b, out_logp);

    // 8) in-place log-softmax over V
    logsoftmax_rows<<<Bdim, 256, 0, stream>>>(out_logp);
}
